// Head_32667521253713
// MI455X (gfx1250) — matrix operand order repaired and checked by reading, not compiled, not run on hardware
//
#include <hip/hip_runtime.h>
#include <hip/hip_bf16.h>

// Single-head causal attention for MI455X (gfx1250, wave32, WMMA).
// Pass 1: project x (fp32) -> Qh, Kh (f16, [B,N,64]) and Vt (f16, [B,64,N]).
//         Weight chunks are staged into LDS *transposed* via the CDNA5 async
//         global->LDS DMA path (GLOBAL_LOAD_ASYNC_TO_LDS_B32, ASYNCcnt).
// Pass 2: flash-attention per 16-row query tile per wave: S = Q K^T (WMMA),
//         online softmax, O += P V (WMMA via per-wave LDS transpose of P).

typedef __attribute__((ext_vector_type(16))) _Float16 v16h;
typedef __attribute__((ext_vector_type(8)))  _Float16 v8h;
typedef __attribute__((ext_vector_type(8)))  float    v8f;
typedef __attribute__((ext_vector_type(4)))  float    v4f;

typedef __attribute__((address_space(3))) float lds_float;

#define BATCH 4
#define SEQ   4096
#define DMODEL 1024
#define HEAD  64

// ---------------------------------------------------------------------------
// Kernel A: q/k/v projections. A block (8 waves) shares one weight matrix and
// computes 8 consecutive 16x64 output tiles. Per K-chunk (32 wide):
//   * all 256 threads async-DMA the 32x64 fp32 W chunk into LDS, transposed
//     to [col][kk] (per-lane scatter -> transpose is free, no VGPR traffic)
//   * each wave: A-frag from x (fp32->f16), B-frags as contiguous LDS b128
//     reads, 4 WMMAs.
// ---------------------------------------------------------------------------
__global__ __launch_bounds__(256) void proj_kernel(
    const float* __restrict__ x,
    const float* __restrict__ Wq, const float* __restrict__ Wk,
    const float* __restrict__ Wv,
    _Float16* __restrict__ Qh, _Float16* __restrict__ Kh,
    _Float16* __restrict__ Vt)
{
    // Transposed W chunk: [col 0..63][kk 0..31], padded row (36 floats = 144B,
    // 16B-aligned rows, banks spread).
    __shared__ __align__(16) float Wlds[64][36];

    const int tid  = threadIdx.x;
    const int lane = tid & 31;
    const int wave = tid >> 5;

    int blk = blockIdx.x;                 // 0 .. 3*4*32-1
    const int w   = blk % 3;              // 0=q 1=k 2=v  (uniform per block)
    blk /= 3;
    const int grp = blk % 32;
    const int b   = blk / 32;
    const int tile = grp * 8 + wave;      // this wave's 16-row tile

    const float* W = (w == 0) ? Wq : (w == 1) ? Wk : Wv;

    const int mr    = tile * 16;
    const int half  = lane >> 4;          // 0 | 1
    const int l15   = lane & 15;
    const int base8 = half * 8;

    // Fill assignment: thread owns 8 consecutive floats of the 32x64 chunk.
    const int fkk  = tid >> 3;            // row within chunk (0..31)
    const int fcol = (tid & 7) * 8;       // first column (0..56)

    const float* xrow = x + ((size_t)b * SEQ + mr + l15) * DMODEL;

    v8f acc[4] = {};

    for (int c = 0; c < 32; ++c) {
        // ---- async DMA fill of the transposed W chunk ----
        const float* gsrc = W + ((size_t)(c * 32 + fkk)) * HEAD + fcol;
#pragma unroll
        for (int i = 0; i < 8; ++i) {
            lds_float* lp = (lds_float*)&Wlds[fcol + i][fkk];
            const float* gp = gsrc + i;
            asm volatile("global_load_async_to_lds_b32 %0, %1, off"
                         :: "v"(lp), "v"(gp) : "memory");
        }
        asm volatile("s_wait_asynccnt 0x0" ::: "memory");
        __syncthreads();

        // ---- A fragment (16x32 f16) from x ----
        const float* xp = xrow + c * 32 + half * 16;
        v4f f0 = *(const v4f*)(xp + 0);
        v4f f1 = *(const v4f*)(xp + 4);
        v4f f2 = *(const v4f*)(xp + 8);
        v4f f3 = *(const v4f*)(xp + 12);
        v16h a;
#pragma unroll
        for (int i = 0; i < 4; ++i) {
            a[i]      = (_Float16)f0[i];
            a[4 + i]  = (_Float16)f1[i];
            a[8 + i]  = (_Float16)f2[i];
            a[12 + i] = (_Float16)f3[i];
        }

        // ---- B fragments from LDS (contiguous 64B per lane) + WMMA ----
#pragma unroll
        for (int t = 0; t < 4; ++t) {
            const float* wl = &Wlds[t * 16 + l15][half * 16];
            v4f w0 = *(const v4f*)(wl + 0);
            v4f w1 = *(const v4f*)(wl + 4);
            v4f w2 = *(const v4f*)(wl + 8);
            v4f w3 = *(const v4f*)(wl + 12);
            v16h bm;
#pragma unroll
            for (int i = 0; i < 4; ++i) {
                bm[i]      = (_Float16)w0[i];
                bm[4 + i]  = (_Float16)w1[i];
                bm[8 + i]  = (_Float16)w2[i];
                bm[12 + i] = (_Float16)w3[i];
            }
            acc[t] = __builtin_amdgcn_wmma_f32_16x16x32_f16(
                false, a, false, bm, (short)0, acc[t], false, false);
        }
        __syncthreads();   // protect Wlds before next chunk's fill
    }

    // Store per C/D layout: VGPR r, lane -> row = r + 8*half, col = t*16 + l15.
#pragma unroll
    for (int t = 0; t < 4; ++t) {
#pragma unroll
        for (int r = 0; r < 8; ++r) {
            const int row = mr + r + 8 * half;
            const int h   = t * 16 + l15;
            const _Float16 vv = (_Float16)acc[t][r];
            if (w == 0)      Qh[((size_t)b * SEQ + row) * HEAD + h] = vv;
            else if (w == 1) Kh[((size_t)b * SEQ + row) * HEAD + h] = vv;
            else             Vt[((size_t)b * HEAD + h) * SEQ + row] = vv; // transposed
        }
    }
}

// ---------------------------------------------------------------------------
// Kernel B: flash attention. One wave owns one 16-row query tile; KV walked in
// 32-row chunks (causal: chunks 0 .. qr/32). 8 WMMA per chunk.
// ---------------------------------------------------------------------------
__global__ __launch_bounds__(256) void attn_kernel(
    const _Float16* __restrict__ Qh, const _Float16* __restrict__ Kh,
    const _Float16* __restrict__ Vt, float* __restrict__ out)
{
    __shared__ _Float16 pbuf[8][16][40];       // per-wave P tile (16x32, padded)

    const int lane = threadIdx.x & 31;
    const int wave = threadIdx.x >> 5;
    int task = blockIdx.x * 8 + wave;          // 0 .. 4*256-1
    const int tile = task % (SEQ / 16);
    const int b    = task / (SEQ / 16);
    const int qr   = tile * 16;
    const int half = lane >> 4;
    const int l15  = lane & 15;
    const int base8 = half * 8;

    // Q A-fragments for head-dim chunks c=0,1 (16x32 each).
    const _Float16* qrow = Qh + ((size_t)b * SEQ + qr + l15) * HEAD;
    v16h aq[2];
#pragma unroll
    for (int c = 0; c < 2; ++c) {
        v8h lo = *(const v8h*)(qrow + c * 32 + half * 16);
        v8h hi = *(const v8h*)(qrow + c * 32 + half * 16 + 8);
#pragma unroll
        for (int i = 0; i < 8; ++i) { aq[c][i] = lo[i]; aq[c][8 + i] = hi[i]; }
    }

    const float NEG_INF = -__builtin_inff();
    const float scale = 0.125f;                 // 1/sqrt(64)
    const float L2E = 1.4426950408889634f;

    float mrow[8], lrow[8];
    v8f accO[4] = {};
#pragma unroll
    for (int r = 0; r < 8; ++r) { mrow[r] = NEG_INF; lrow[r] = 0.0f; }

    const int nchunks = qr / 32 + 1;

    for (int j = 0; j < nchunks; ++j) {
        const int kv0 = j * 32;

        if (j + 1 < nchunks) {  // pull next K/V chunks toward L2/L0
            __builtin_prefetch(Kh + ((size_t)b * SEQ + kv0 + 32 + l15) * HEAD, 0, 1);
            __builtin_prefetch(Vt + ((size_t)b * HEAD + l15) * SEQ + kv0 + 32, 0, 1);
        }

        // ---- S = Q K^T : two 16x16 subtiles (cols kv0.. and kv0+16..) ----
        v8f s[2];
#pragma unroll
        for (int sub = 0; sub < 2; ++sub) {
            v8f sa = {};
            const _Float16* krow =
                Kh + ((size_t)b * SEQ + kv0 + sub * 16 + l15) * HEAD;
#pragma unroll
            for (int c = 0; c < 2; ++c) {
                // B frag: lane = kv-col (l15) + 16*(K/16); 16 contiguous halves.
                v16h bk = *(const v16h*)(krow + c * 32 + half * 16);
                sa = __builtin_amdgcn_wmma_f32_16x16x32_f16(
                    false, aq[c], false, bk, (short)0, sa, false, false);
            }
            s[sub] = sa;
        }

        // ---- scale, mask (causal + tril==0 faithfulness), online softmax ----
#pragma unroll
        for (int r = 0; r < 8; ++r) {
            const int row = qr + r + 8 * half;
            const int c0  = kv0 + l15;
            const int c1  = c0 + 16;
            float s0 = s[0][r] * scale;
            float s1 = s[1][r] * scale;
            if (c0 > row || s0 == 0.0f) s0 = NEG_INF;
            if (c1 > row || s1 == 0.0f) s1 = NEG_INF;

            float vmax = fmaxf(s0, s1);
#pragma unroll
            for (int o = 8; o; o >>= 1) vmax = fmaxf(vmax, __shfl_xor(vmax, o, 32));
            const float newm = fmaxf(mrow[r], vmax);
            const float corr = (newm == NEG_INF)
                                   ? 0.0f
                                   : __builtin_exp2f((mrow[r] - newm) * L2E);
            const float e0 = (s0 == NEG_INF) ? 0.0f
                                             : __builtin_exp2f((s0 - newm) * L2E);
            const float e1 = (s1 == NEG_INF) ? 0.0f
                                             : __builtin_exp2f((s1 - newm) * L2E);
            float psum = e0 + e1;
#pragma unroll
            for (int o = 8; o; o >>= 1) psum += __shfl_xor(psum, o, 32);
            lrow[r] = lrow[r] * corr + psum;
            mrow[r] = newm;
#pragma unroll
            for (int t = 0; t < 4; ++t) accO[t][r] *= corr;

            // C-layout -> LDS (own wave's slice; same-wave DS is in-order).
            pbuf[wave][r + 8 * half][l15]      = (_Float16)e0;
            pbuf[wave][r + 8 * half][16 + l15] = (_Float16)e1;
        }

        // ---- P A-fragment (16x32) from LDS in contiguous k order (k = 16*half + s) ----
        v16h ap;
#pragma unroll
        for (int h = 0; h < 8; ++h) {
            ap[h]     = pbuf[wave][l15][half * 16 + h];
            ap[h + 8] = pbuf[wave][l15][half * 16 + 8 + h];
        }

        // ---- O += P V : V transposed so B frags are contiguous ----
#pragma unroll
        for (int t = 0; t < 4; ++t) {
            const _Float16* vp =
                Vt + ((size_t)b * HEAD + t * 16 + l15) * SEQ + kv0 + half * 16;
            v16h bv = *(const v16h*)vp;
            accO[t] = __builtin_amdgcn_wmma_f32_16x16x32_f16(
                false, ap, false, bv, (short)0, accO[t], false, false);
        }
    }

    // ---- normalize and store fp32 output [B,N,64] ----
#pragma unroll
    for (int t = 0; t < 4; ++t) {
#pragma unroll
        for (int r = 0; r < 8; ++r) {
            const int row = qr + r + 8 * half;
            out[((size_t)b * SEQ + row) * HEAD + t * 16 + l15] =
                accO[t][r] / lrow[r];
        }
    }
}

// ---------------------------------------------------------------------------
extern "C" void kernel_launch(void* const* d_in, const int* in_sizes, int n_in,
                              void* d_out, int out_size, void* d_ws, size_t ws_size,
                              hipStream_t stream)
{
    const float* x  = (const float*)d_in[0];
    const float* Wq = (const float*)d_in[1];
    const float* Wk = (const float*)d_in[2];
    const float* Wv = (const float*)d_in[3];
    float* out = (float*)d_out;

    // Workspace: Qh | Kh | Vt, each B*N*H f16 = 2 MB (6 MB total).
    _Float16* Qh = (_Float16*)d_ws;
    _Float16* Kh = Qh + (size_t)BATCH * SEQ * HEAD;
    _Float16* Vt = Kh + (size_t)BATCH * SEQ * HEAD;

    // Pass 1: 3 matrices * 4 batches * 32 tile-groups = 384 blocks, 8 waves each.
    proj_kernel<<<3 * BATCH * (SEQ / 16 / 8), 256, 0, stream>>>(
        x, Wq, Wk, Wv, Qh, Kh, Vt);

    // Pass 2: 4 * 256 = 1024 wave-tasks, 8 waves/block.
    attn_kernel<<<(BATCH * (SEQ / 16)) / 8, 256, 0, stream>>>(Qh, Kh, Vt, out);
}
